// GCN_pyg_17119739641949
// MI455X (gfx1250) — matrix operand order repaired and checked by reading, not compiled, not run on hardware
//
#include <hip/hip_runtime.h>
#include <hip/hip_bf16.h>
#include <math.h>

typedef __attribute__((ext_vector_type(16))) __bf16 v16bf;
typedef __attribute__((ext_vector_type(8)))  float  v8f;

// round-to-nearest-even fp32 -> bf16 (bit-exact, no reliance on cast support)
__device__ __forceinline__ __bf16 f2bf(float f) {
  union { float f; unsigned u; } x; x.f = f;
  unsigned r = x.u + 0x7FFFu + ((x.u >> 16) & 1u);
  unsigned short h = (unsigned short)(r >> 16);
  __bf16 o; __builtin_memcpy(&o, &h, 2); return o;
}

__device__ __forceinline__ void atomAddF32(float* p, float v) {
  (void)__hip_atomic_fetch_add(p, v, __ATOMIC_RELAXED, __HIP_MEMORY_SCOPE_AGENT);
}

__global__ void k_fill(float* __restrict__ p, float v, long long n) {
  long long i = (long long)blockIdx.x * blockDim.x + threadIdx.x;
  if (i < n) p[i] = v;
}

__global__ void k_deg_acc(const int* __restrict__ col, float* __restrict__ deg, int E) {
  int i = blockIdx.x * blockDim.x + threadIdx.x;
  if (i < E) atomAddF32(&deg[col[i]], 1.0f);
}

__global__ void k_dinv(float* __restrict__ deg, int n) {
  int i = blockIdx.x * blockDim.x + threadIdx.x;
  if (i < n) { float d = deg[i]; deg[i] = d > 0.f ? rsqrtf(d) : 0.f; }
}

// Pack row-major f32 [M x K] into WMMA bf16 A-fragment layout.
// A 16x32 bf16 fragment (ISA 7.12.2): lane l, element i holds
//   m = 16*mt + (l & 15),  k = 32*kt + (i & 7) + ((i >> 3) << 4) + ((l >> 4) << 3)
// Fragment (mt,kt) stored flat at dst + (mt*KT + kt)*512 + l*16 + i  -> contiguous loads in GEMM.
// Optionally fuses +bias and ReLU (for GEMM2's input = relu(agg1 + b1)).
__global__ void k_pack_a(const float* __restrict__ src, const float* __restrict__ bias,
                         int reluFlag, __bf16* __restrict__ dst, int K, int KT, long long total) {
  long long id = (long long)blockIdx.x * blockDim.x + threadIdx.x;
  if (id >= total) return;
  int i    = (int)(id & 15);
  int lane = (int)((id >> 4) & 31);
  long long frag = id >> 9;
  int kt = (int)(frag % KT);
  long long mt = frag / KT;
  long long m = mt * 16 + (lane & 15);
  int k = kt * 32 + (i & 7) + ((i >> 3) << 4) + ((lane >> 4) << 3);
  float v = src[m * K + k];
  if (bias) v += bias[k];
  if (reluFlag) v = fmaxf(v, 0.f);
  dst[id] = f2bf(v);
}

// Pack W [K x N] row-major f32 into WMMA bf16 B-fragment layout.
// B 32x16 bf16 fragment: lane l, element i holds k = 32*kt + (i&7) + ((i>>3)<<4) + ((l>>4)<<3), n = 16*nt + (l&15).
// Fragment (nt,kt) at dst + (nt*KT + kt)*512 + l*16 + i.
__global__ void k_pack_b(const float* __restrict__ W, __bf16* __restrict__ Wp,
                         int N, int KT, int total) {
  int id = blockIdx.x * blockDim.x + threadIdx.x;
  if (id >= total) return;
  int i = id & 15, lane = (id >> 4) & 31;
  int frag = id >> 9;
  int kt = frag % KT, nt = frag / KT;
  int k = kt * 32 + (i & 7) + ((i >> 3) << 4) + ((lane >> 4) << 3);
  int n = nt * 16 + (lane & 15);
  Wp[id] = f2bf(W[(long long)k * N + n]);
}

// One 16x16 output tile per wave; K-loop of v_wmma_f32_16x16x32_bf16.
__global__ void __launch_bounds__(256)
k_gemm_wmma(const __bf16* __restrict__ Ap, const __bf16* __restrict__ Bp,
            float* __restrict__ C, int MT, int NT, int KT, int N) {
  int wg = blockIdx.x * (blockDim.x >> 5) + (threadIdx.x >> 5);
  if (wg >= MT * NT) return;                 // wave-uniform: EXEC stays all-ones
  int lane = threadIdx.x & 31;
  int mtile = wg / NT, ntile = wg % NT;
  const __bf16* ap = Ap + (long long)mtile * KT * 512 + lane * 16;
  const __bf16* bp = Bp + (long long)ntile * KT * 512 + lane * 16;
  v8f acc = {0.f, 0.f, 0.f, 0.f, 0.f, 0.f, 0.f, 0.f};
  for (int kt = 0; kt < KT; ++kt) {
    v16bf a = *(const v16bf*)(ap + (long long)kt * 512);
    v16bf b = *(const v16bf*)(bp + (long long)kt * 512);
    acc = __builtin_amdgcn_wmma_f32_16x16x32_bf16(false, a, false, b, (short)0, acc,
                                                  false, false);
  }
  // C/D layout: VGPR j -> m = 8*(lane>=16) + j, n = lane&15
  int n = ntile * 16 + (lane & 15);
  long long mbase = (long long)mtile * 16 + ((lane >> 4) << 3);
#pragma unroll
  for (int j = 0; j < 8; ++j) C[(mbase + j) * N + n] = acc[j];
}

// msg = h[row]*norm ; agg[col] += msg  (edges, then implicit self-loops).
// One float4 per thread; L2-resident fp32 atomics. F = 4 << fshift.
__global__ void k_scatter(const float* __restrict__ h, const int* __restrict__ row,
                          const int* __restrict__ col, const float* __restrict__ dinv,
                          float* __restrict__ agg, int E, int nNodes, int fshift) {
  long long total = ((long long)E + nNodes) << fshift;
  long long t = (long long)blockIdx.x * blockDim.x + threadIdx.x;
  if (t >= total) return;
  long long e = t >> fshift;
  int f4 = (int)(t & ((1 << fshift) - 1));
  int r, c;
  if (e < E) { r = row[e]; c = col[e]; } else { int s = (int)(e - E); r = s; c = s; }
  float w = dinv[r] * dinv[c];
  int F = 4 << fshift;
  float4 hv = ((const float4*)(h + (long long)r * F))[f4];
  float* dst = agg + (long long)c * F + 4 * f4;
  atomAddF32(dst + 0, hv.x * w);
  atomAddF32(dst + 1, hv.y * w);
  atomAddF32(dst + 2, hv.z * w);
  atomAddF32(dst + 3, hv.w * w);
}

// out[p] = sigmoid( (h[i1]+b2)·Wl[0:F] + (h[i2]+b2)·Wl[F:2F] + bl ). One wave32 per pair.
__global__ void k_pair(const float* __restrict__ h, const int* __restrict__ mask,
                       const float* __restrict__ b2, const float* __restrict__ Wl,
                       const float* __restrict__ bl, float* __restrict__ out,
                       int P, int F) {
  int wave = (blockIdx.x * blockDim.x + threadIdx.x) >> 5;
  int lane = threadIdx.x & 31;
  if (wave >= P) return;
  int i1 = mask[2 * wave], i2 = mask[2 * wave + 1];
  float s = 0.f;
  for (int j = lane; j < F; j += 32) {
    float bj = b2[j];
    s += (h[(long long)i1 * F + j] + bj) * Wl[j];
    s += (h[(long long)i2 * F + j] + bj) * Wl[F + j];
  }
  for (int off = 16; off; off >>= 1) s += __shfl_xor(s, off, 32);
  if (lane == 0) out[wave] = 1.f / (1.f + __expf(-(s + bl[0])));
}

extern "C" void kernel_launch(void* const* d_in, const int* in_sizes, int n_in,
                              void* d_out, int out_size, void* d_ws, size_t ws_size,
                              hipStream_t stream) {
  const int*   edge = (const int*)d_in[0];     // [2, E]
  const float* X    = (const float*)d_in[1];   // [N, 128]
  const int*   mask = (const int*)d_in[2];     // [P, 2]
  const float* W1   = (const float*)d_in[3];   // [128, 256]
  const float* b1   = (const float*)d_in[4];   // [256]
  const float* W2   = (const float*)d_in[5];   // [256, 128]
  const float* b2   = (const float*)d_in[6];   // [128]
  const float* Wl   = (const float*)d_in[7];   // [256]
  const float* bl   = (const float*)d_in[8];   // [1]
  float* out = (float*)d_out;

  const int E  = in_sizes[0] / 2;
  const int Nn = in_sizes[1] / 128;   // 50000
  const int P  = in_sizes[2] / 2;     // 100000
  const int F1 = 128, F2 = 256;
  const int MT = Nn / 16;             // 3125 (50000 divisible by 16)

  const int* row = edge;
  const int* col = edge + E;

  // workspace layout (~116 MB, regions reused across phases)
  char* ws = (char*)d_ws;
  size_t off = 0;
  auto alloc = [&](size_t bytes) -> char* {
    char* p = ws + off;
    off = (off + bytes + 255) & ~(size_t)255;
    return p;
  };
  float*  deg  = (float*)alloc((size_t)Nn * 4);            // deg -> dinv in place
  __bf16* Ap1  = (__bf16*)alloc((size_t)Nn * F1 * 2);      // packed X
  __bf16* W1p  = (__bf16*)alloc((size_t)F1 * F2 * 2);
  __bf16* W2p  = (__bf16*)alloc((size_t)F2 * F1 * 2);
  char*   regA = alloc((size_t)Nn * F2 * 4);               // h1, then a1p|h2
  char*   regB = alloc((size_t)Nn * F2 * 4);               // agg1, then agg2
  float*  h1   = (float*)regA;
  float*  agg1 = (float*)regB;
  __bf16* a1p  = (__bf16*)regA;                            // 25.6MB (first half)
  float*  h2   = (float*)(regA + (size_t)Nn * F2 * 2);     // 25.6MB (second half)
  float*  agg2 = (float*)regB;

  const int B = 256;

  // ---- gcn_norm: deg init 1.0 bakes in self-loops; dinv computed in place
  k_fill<<<(Nn + B - 1) / B, B, 0, stream>>>(deg, 1.0f, Nn);
  k_deg_acc<<<(E + B - 1) / B, B, 0, stream>>>(col, deg, E);
  k_dinv<<<(Nn + B - 1) / B, B, 0, stream>>>(deg, Nn);

  // ---- pack operands for WMMA
  long long tA1 = (long long)Nn * F1;
  k_pack_a<<<(unsigned)((tA1 + B - 1) / B), B, 0, stream>>>(X, nullptr, 0, Ap1, F1, F1 / 32, tA1);
  k_pack_b<<<(F1 * F2 + B - 1) / B, B, 0, stream>>>(W1, W1p, F2, F1 / 32, F1 * F2);
  k_pack_b<<<(F2 * F1 + B - 1) / B, B, 0, stream>>>(W2, W2p, F1, F2 / 32, F2 * F1);

  // ---- GEMM1: h1 = X @ W1  [Nn x 256]
  int NT1 = F2 / 16;
  k_gemm_wmma<<<(MT * NT1 + 7) / 8, B, 0, stream>>>(Ap1, W1p, h1, MT, NT1, F1 / 32, F2);

  // ---- scatter 1 (L2-resident atomics)
  long long tz1 = (long long)Nn * F2;
  k_fill<<<(unsigned)((tz1 + B - 1) / B), B, 0, stream>>>(agg1, 0.f, tz1);
  long long ts1 = ((long long)E + Nn) << 6;   // F2/4 = 64 chunks/edge
  k_scatter<<<(unsigned)((ts1 + B - 1) / B), B, 0, stream>>>(h1, row, col, deg, agg1, E, Nn, 6);

  // ---- bias + ReLU fused into A-fragment pack for GEMM2
  long long tA2 = (long long)Nn * F2;
  k_pack_a<<<(unsigned)((tA2 + B - 1) / B), B, 0, stream>>>(agg1, b1, 1, a1p, F2, F2 / 32, tA2);

  // ---- GEMM2: h2 = relu(agg1+b1) @ W2  [Nn x 128]
  int NT2 = F1 / 16;
  k_gemm_wmma<<<(MT * NT2 + 7) / 8, B, 0, stream>>>(a1p, W2p, h2, MT, NT2, F2 / 32, F1);

  // ---- scatter 2
  long long tz2 = (long long)Nn * F1;
  k_fill<<<(unsigned)((tz2 + B - 1) / B), B, 0, stream>>>(agg2, 0.f, tz2);
  long long ts2 = ((long long)E + Nn) << 5;   // F1/4 = 32 chunks/edge
  k_scatter<<<(unsigned)((ts2 + B - 1) / B), B, 0, stream>>>(h2, row, col, deg, agg2, E, Nn, 5);

  // ---- pair head: fuses +b2, concat-dot with Wl, +bl, sigmoid
  k_pair<<<(P + 7) / 8, B, 0, stream>>>(agg2, mask, b2, Wl, bl, out, P, F1);

  (void)n_in; (void)out_size; (void)ws_size;
}
